// Simulator_86088324481852
// MI455X (gfx1250) — matrix operand order repaired and checked by reading, not compiled, not run on hardware
//
#include <hip/hip_runtime.h>

// ---------------------------------------------------------------------------
// Types for CDNA5 WMMA (wave32)
// ---------------------------------------------------------------------------
typedef __attribute__((ext_vector_type(16))) _Float16 v16h;
typedef __attribute__((ext_vector_type(8)))  _Float16 v8h;
typedef __attribute__((ext_vector_type(8)))  float    v8f;

#define N_PART 20000
#define N_EDGE 320000
#define LATENT 128
#define NSTEPS 10

// ---------------------------------------------------------------------------
// Async global->LDS copy (CDNA5 GLOBAL_LOAD_ASYNC_TO_LDS_B128, ASYNCcnt).
// Builtin signature (from hipcc diagnostic): param0 is
// 'int __attribute__((vector_size(16))) __device__ *'  i.e. v4i in AS(1);
// param1 is the LDS-side (AS(3)) pointer; then imm offset + cpol.
// ---------------------------------------------------------------------------
#if defined(__has_builtin)
#if __has_builtin(__builtin_amdgcn_global_load_async_to_lds_b128) && \
    __has_builtin(__builtin_amdgcn_s_wait_asynccnt)
#define HAVE_ASYNC_LDS 1
#endif
#endif

#ifdef HAVE_ASYNC_LDS
typedef int v4i_vs __attribute__((vector_size(16)));
typedef __attribute__((address_space(1))) v4i_vs* g_v4p;
typedef __attribute__((address_space(3))) v4i_vs* l_v4p;
__device__ __forceinline__ void async_copy16(const void* g, void* l) {
  __builtin_amdgcn_global_load_async_to_lds_b128((g_v4p)g, (l_v4p)l, 0, 0);
}
#endif

// ---------------------------------------------------------------------------
// Fragment loaders (wave32 WMMA; A/B use this code's contiguous k order, same for both)
// A 16x32 f16: lanes 0-15 row=l, K=0..15; lanes 16-31 row=l-16,
//              K=16..31  -> one contiguous 32B chunk (two 16B loads).
// B 32x16 f16: lanes 0-15 col=l, K=0..15; lanes 16-31 col=l-16, K=16..31
//              -> one contiguous 32B chunk (two 16B loads).
// C/D 16x16 f32: VGPR r, lanes 0-15 row=r col=l; lanes 16-31 row=r+8 col=l-16.
// ---------------------------------------------------------------------------
__device__ __forceinline__ v16h cat16(v8h lo, v8h hi) {
  return __builtin_shufflevector(lo, hi, 0, 1, 2, 3, 4, 5, 6, 7,
                                 8, 9, 10, 11, 12, 13, 14, 15);
}

__device__ __forceinline__ v16h load_a_frag(const _Float16* lds, int rowBase,
                                            int stride, int k0, int lane) {
  int row = rowBase + (lane & 15);
  int kb = (lane < 16) ? 0 : 16;
  const v8h* p0 = (const v8h*)(lds + row * stride + k0 + kb);
  return cat16(p0[0], p0[1]);
}

__device__ __forceinline__ v16h load_b_frag(const _Float16* wt, int n0,
                                            int stride, int k0, int lane) {
  int n = n0 + (lane & 15);
  int kb = (lane < 16) ? 0 : 16;
  const v8h* p = (const v8h*)(wt + (size_t)n * stride + k0 + kb);
  return cat16(p[0], p[1]);
}

__device__ __forceinline__ v8h pack8(const float* f) {
  v8h r;
#pragma unroll
  for (int j = 0; j < 8; ++j) r[j] = (_Float16)f[j];
  return r;
}

// One 32xKK @ KKx128 layer: this wave produces two 16x16 C tiles (rows 0-15
// and 16-31 of the block) for its 16-column stripe n0..n0+15.
template <int KK>
__device__ __forceinline__ void gemm_2tiles(const _Float16* ldsIn,
                                            const _Float16* wt, int n0,
                                            int lane, v8f& c0, v8f& c1) {
  c0 = (v8f){};
  c1 = (v8f){};
#pragma unroll
  for (int k0 = 0; k0 < KK; k0 += 32) {
    v16h B  = load_b_frag(wt, n0, KK, k0, lane);
    v16h A0 = load_a_frag(ldsIn, 0, KK, k0, lane);
    v16h A1 = load_a_frag(ldsIn, 16, KK, k0, lane);
    c0 = __builtin_amdgcn_wmma_f32_16x16x32_f16(false, A0, false, B, (short)0,
                                                c0, false, false);
    c1 = __builtin_amdgcn_wmma_f32_16x16x32_f16(false, A1, false, B, (short)0,
                                                c1, false, false);
  }
}

__device__ __forceinline__ void epilogue_hidden(v8f c, float bval,
                                                _Float16* ldsOut, int mrowBase,
                                                int n0, int lane) {
  int col = n0 + (lane & 15);
  int radd = (lane >= 16) ? 8 : 0;
#pragma unroll
  for (int r = 0; r < 8; ++r) {
    float h = c[r] + bval;
    h = h > 0.0f ? h : 0.0f;  // ReLU
    ldsOut[(mrowBase + r + radd) * LATENT + col] = (_Float16)h;
  }
}

__device__ __forceinline__ void epilogue_final(v8f c, float bval, float* ldsOut,
                                               int mrowBase, int n0, int lane,
                                               bool relu) {
  int col = n0 + (lane & 15);
  int radd = (lane >= 16) ? 8 : 0;
#pragma unroll
  for (int r = 0; r < 8; ++r) {
    float h = c[r] + bval;
    if (relu) h = h > 0.0f ? h : 0.0f;
    ldsOut[(mrowBase + r + radd) * LATENT + col] = h;
  }
}

// ---------------------------------------------------------------------------
// Fused 2/3-layer MLP block kernel. 256 threads = 8 waves; 32 rows per block.
// MODE: 0 enc-node (X buf), 1 enc-edge (X buf), 2 proc-edge (gather e|v[s]|v[r],
//       residual e += y, atomic agg[r] += y), 3 proc-node (v|agg, residual),
//       4 decoder first two layers (relu final, no LN).
// ---------------------------------------------------------------------------
struct MlpArgs {
  const _Float16* X;
  const _Float16* Wt1;
  const _Float16* Wt2;
  const _Float16* Wt3;
  const float* b1;
  const float* b2;
  const float* b3;
  const float* ln_g;
  const float* ln_b;
  _Float16* out;
  const _Float16* e_in;
  const _Float16* v_in;
  const float* agg_in;
  float* agg_out;
  const int* s_idx;
  const int* r_idx;
};

template <int K0PAD, int NLAYERS, int MODE>
__global__ __launch_bounds__(256) void mlp_block_kernel(MlpArgs a) {
  __shared__ _Float16 sX[32 * K0PAD];
  __shared__ _Float16 sA[32 * LATENT];
  __shared__ _Float16 sB[32 * LATENT];
  __shared__ float sO[32 * LATENT];
  __shared__ float sMu[32];
  __shared__ float sRs[32];
  __shared__ float sSum[256];
  __shared__ float sSq[256];

  const int tid = threadIdx.x;
  const int lane = tid & 31;
  const int wv = tid >> 5;
  const int n0 = wv * 16;
  const long base = (long)blockIdx.x * 32;

  // ---- stage input block (32 x K0PAD f16) into LDS -------------------------
  constexpr int C8 = K0PAD / 8;
  for (int idx = tid; idx < 32 * C8; idx += 256) {
    int row = idx / C8;
    int col = (idx - row * C8) * 8;
    long grow = base + row;
    _Float16* dst = sX + row * K0PAD + col;
    if (MODE == 3 && col >= 128) {
      // agg is f32: convert through VGPRs
      const float* p = a.agg_in + grow * LATENT + (col - 128);
      v8h val;
#pragma unroll
      for (int j = 0; j < 8; ++j) val[j] = (_Float16)p[j];
      *(v8h*)dst = val;
      continue;
    }
    const _Float16* src;
    if (MODE == 2) {  // [e | v[s] | v[r]]
      if (col < 128)
        src = a.e_in + grow * LATENT + col;
      else if (col < 256)
        src = a.v_in + (long)a.s_idx[grow] * LATENT + (col - 128);
      else
        src = a.v_in + (long)a.r_idx[grow] * LATENT + (col - 256);
    } else if (MODE == 3) {
      src = a.v_in + grow * LATENT + col;
    } else {  // dense X buffer with stride K0PAD
      src = a.X + grow * K0PAD + col;
    }
#ifdef HAVE_ASYNC_LDS
    async_copy16(src, dst);  // global -> LDS, no VGPR round-trip
#else
    *(v8h*)dst = *(const v8h*)src;
#endif
  }
#ifdef HAVE_ASYNC_LDS
  __builtin_amdgcn_s_wait_asynccnt(0);
#endif
  __syncthreads();

  v8f c0, c1;
  // ---- layer 1 -------------------------------------------------------------
  gemm_2tiles<K0PAD>(sX, a.Wt1, n0, lane, c0, c1);
  {
    float bv = a.b1[n0 + (lane & 15)];
    epilogue_hidden(c0, bv, sA, 0, n0, lane);
    epilogue_hidden(c1, bv, sA, 16, n0, lane);
  }
  __syncthreads();

  if (NLAYERS == 3) {
    // ---- layer 2 (hidden) --------------------------------------------------
    gemm_2tiles<LATENT>(sA, a.Wt2, n0, lane, c0, c1);
    {
      float bv = a.b2[n0 + (lane & 15)];
      epilogue_hidden(c0, bv, sB, 0, n0, lane);
      epilogue_hidden(c1, bv, sB, 16, n0, lane);
    }
    __syncthreads();
    // ---- layer 3 (final, f32) ---------------------------------------------
    gemm_2tiles<LATENT>(sB, a.Wt3, n0, lane, c0, c1);
    {
      float bv = a.b3[n0 + (lane & 15)];
      epilogue_final(c0, bv, sO, 0, n0, lane, false);
      epilogue_final(c1, bv, sO, 16, n0, lane, false);
    }
  } else {
    // ---- layer 2 (final, relu for decoder hidden) -------------------------
    gemm_2tiles<LATENT>(sA, a.Wt2, n0, lane, c0, c1);
    {
      float bv = a.b2[n0 + (lane & 15)];
      epilogue_final(c0, bv, sO, 0, n0, lane, true);
      epilogue_final(c1, bv, sO, 16, n0, lane, true);
    }
  }
  __syncthreads();

  // ---- LN + writeout: thread owns 16 contiguous cols of one row -----------
  constexpr bool LN = (MODE != 4);
  const int row = tid >> 3;
  const int cb = (tid & 7) * 16;
  const long grow = base + row;

  float t[16];
#pragma unroll
  for (int j = 0; j < 16; ++j) t[j] = sO[row * LATENT + cb + j];

  if (LN) {
    float s = 0.0f, ss = 0.0f;
#pragma unroll
    for (int j = 0; j < 16; ++j) {
      s += t[j];
      ss += t[j] * t[j];
    }
    sSum[tid] = s;
    sSq[tid] = ss;
    __syncthreads();
    if (tid < 32) {
      float rs = 0.0f, rss = 0.0f;
#pragma unroll
      for (int j = 0; j < 8; ++j) {
        rs += sSum[tid * 8 + j];
        rss += sSq[tid * 8 + j];
      }
      float mu = rs * (1.0f / LATENT);
      float var = rss * (1.0f / LATENT) - mu * mu;
      sMu[tid] = mu;
      sRs[tid] = rsqrtf(var + 1e-5f);
    }
    __syncthreads();
    float mu = sMu[row];
    float rs = sRs[row];
#pragma unroll
    for (int j = 0; j < 16; ++j)
      t[j] = (t[j] - mu) * rs * a.ln_g[cb + j] + a.ln_b[cb + j];
  }

  if (MODE == 2 || MODE == 3) {
    // residual: out += y (vectorized f16 RMW, two 16B chunks)
    _Float16* o = a.out + grow * LATENT + cb;
    v8h old0 = *(const v8h*)o;
    v8h old1 = *(const v8h*)(o + 8);
    float r0[8], r1[8];
#pragma unroll
    for (int j = 0; j < 8; ++j) {
      r0[j] = (float)old0[j] + t[j];
      r1[j] = (float)old1[j] + t[8 + j];
    }
    *(v8h*)o = pack8(r0);
    *(v8h*)(o + 8) = pack8(r1);
    if (MODE == 2) {
      // segment_sum scatter (f32 hardware atomics)
      float* ag = a.agg_out + (long)a.r_idx[grow] * LATENT + cb;
#pragma unroll
      for (int j = 0; j < 16; ++j) unsafeAtomicAdd(ag + j, t[j]);
    }
  } else {
    _Float16* o = a.out + grow * LATENT + cb;
    *(v8h*)o = pack8(t);
    *(v8h*)(o + 8) = pack8(t + 8);
  }
}

// ---------------------------------------------------------------------------
// Weight prep: f32 [K][N] -> f16 [N][Kpad] (transposed, zero-padded K)
// ---------------------------------------------------------------------------
__global__ void transpose_w_kernel(const float* __restrict__ W,
                                   _Float16* __restrict__ Wt, int K, int Kpad,
                                   int Nout) {
  int idx = blockIdx.x * 256 + threadIdx.x;
  int total = Nout * Kpad;
  if (idx >= total) return;
  int n = idx / Kpad;
  int k = idx - n * Kpad;
  float w = (k < K) ? W[k * Nout + n] : 0.0f;
  Wt[idx] = (_Float16)w;
}

__global__ void zero_f32_kernel(float* __restrict__ p, long n) {
  long i = (long)blockIdx.x * 256 + threadIdx.x;
  if (i < n) p[i] = 0.0f;
}

// ---------------------------------------------------------------------------
// Feature construction
// ---------------------------------------------------------------------------
__global__ void node_features_kernel(const float* __restrict__ pos,
                                     const int* __restrict__ types,
                                     const float* __restrict__ nmean,
                                     const float* __restrict__ nstd,
                                     const float* __restrict__ embed,
                                     _Float16* __restrict__ out, int n) {
  int i = blockIdx.x * 256 + threadIdx.x;
  if (i >= n) return;
  const float* p = pos + (long)i * 12;
  _Float16* o = out + (long)i * 32;
  float raw[14];
#pragma unroll
  for (int j = 0; j < 5; ++j) {
    raw[j * 2 + 0] = p[(j + 1) * 2 + 0] - p[j * 2 + 0];
    raw[j * 2 + 1] = p[(j + 1) * 2 + 1] - p[j * 2 + 1];
  }
  const float invR = 1.0f / 0.015f;
  float c0 = p[10], c1 = p[11];
  raw[10] = fminf(fmaxf(c0 * invR, -1.0f), 1.0f);
  raw[11] = fminf(fmaxf(c1 * invR, -1.0f), 1.0f);
  raw[12] = fminf(fmaxf((1.0f - c0) * invR, -1.0f), 1.0f);
  raw[13] = fminf(fmaxf((1.0f - c1) * invR, -1.0f), 1.0f);
#pragma unroll
  for (int j = 0; j < 14; ++j)
    o[j] = (_Float16)((raw[j] - nmean[j]) / nstd[j]);
  int t = types[i];
#pragma unroll
  for (int k = 0; k < 16; ++k) o[14 + k] = (_Float16)embed[t * 16 + k];
  o[30] = (_Float16)0.0f;
  o[31] = (_Float16)0.0f;
}

__global__ void edge_features_kernel(const float* __restrict__ pos,
                                     const int* __restrict__ sidx,
                                     const int* __restrict__ ridx,
                                     const float* __restrict__ emean,
                                     const float* __restrict__ estd,
                                     _Float16* __restrict__ out, int n) {
  int i = blockIdx.x * 256 + threadIdx.x;
  if (i >= n) return;
  const float invR = 1.0f / 0.015f;
  long s = sidx[i], r = ridx[i];
  float rel0 = (pos[s * 12 + 10] - pos[r * 12 + 10]) * invR;
  float rel1 = (pos[s * 12 + 11] - pos[r * 12 + 11]) * invR;
  float dist = sqrtf(rel0 * rel0 + rel1 * rel1);
  _Float16* o = out + (long)i * 32;
  o[0] = (_Float16)((rel0 - emean[0]) / estd[0]);
  o[1] = (_Float16)((rel1 - emean[1]) / estd[1]);
  o[2] = (_Float16)((dist - emean[2]) / estd[2]);
#pragma unroll
  for (int j = 3; j < 32; ++j) o[j] = (_Float16)0.0f;
}

// ---------------------------------------------------------------------------
// Decoder tail: h2 (f16 [N][128]) @ W2 (f32 [128][2]) + b2, then de-normalize.
// ---------------------------------------------------------------------------
__global__ void dec_final_kernel(const _Float16* __restrict__ h2,
                                 const float* __restrict__ W2,
                                 const float* __restrict__ b2,
                                 const float* __restrict__ ostd,
                                 const float* __restrict__ omean,
                                 float* __restrict__ out, int n) {
  int i = blockIdx.x * 256 + threadIdx.x;
  if (i >= n) return;
  const _Float16* h = h2 + (long)i * LATENT;
  float acc0 = b2[0], acc1 = b2[1];
#pragma unroll 8
  for (int k = 0; k < LATENT; ++k) {
    float x = (float)h[k];
    acc0 += x * W2[k * 2 + 0];
    acc1 += x * W2[k * 2 + 1];
  }
  out[i * 2 + 0] = acc0 * ostd[0] + omean[0];
  out[i * 2 + 1] = acc1 * ostd[1] + omean[1];
}

// ---------------------------------------------------------------------------
// Host side
// ---------------------------------------------------------------------------
// Input index map (jax pytree flatten, dict keys sorted):
enum {
  I_POS = 0, I_TYPES = 1, I_EIDX = 2,
  I_DEC_W0 = 3, I_DEC_W1 = 4, I_DEC_W2 = 5, I_DEC_B0 = 6, I_DEC_B1 = 7,
  I_DEC_B2 = 8,
  I_EDGE_MEAN = 9, I_EDGE_STD = 10, I_EMBED = 11,
  I_EE_W0 = 12, I_EE_W1 = 13, I_EE_W2 = 14, I_EE_B0 = 15, I_EE_B1 = 16,
  I_EE_B2 = 17, I_EE_LNB = 18, I_EE_LNG = 19,
  I_EN_W0 = 20, I_EN_W1 = 21, I_EN_W2 = 22, I_EN_B0 = 23, I_EN_B1 = 24,
  I_EN_B2 = 25, I_EN_LNB = 26, I_EN_LNG = 27,
  I_NODE_MEAN = 28, I_NODE_STD = 29, I_OUT_MEAN = 30, I_OUT_STD = 31,
  I_PROC = 32  // 10 steps x (edge: W0,W1,W2,b0,b1,b2,ln_b,ln_g ; node: same)
};

extern "C" void kernel_launch(void* const* d_in, const int* in_sizes, int n_in,
                              void* d_out, int out_size, void* d_ws,
                              size_t ws_size, hipStream_t stream) {
  const float* pos = (const float*)d_in[I_POS];
  const int* types = (const int*)d_in[I_TYPES];
  const int* sidx = (const int*)d_in[I_EIDX];
  const int* ridx = sidx + N_EDGE;
  auto F = [&](int i) { return (const float*)d_in[i]; };

  // bump allocator over workspace
  size_t off = 0;
  auto alloc = [&](size_t bytes) -> void* {
    size_t a = (off + 255) & ~(size_t)255;
    off = a + bytes;
    return (void*)((char*)d_ws + a);
  };
  auto allocH = [&](size_t n) { return (_Float16*)alloc(n * 2); };
  auto allocF = [&](size_t n) { return (float*)alloc(n * 4); };

  // transposed/padded f16 weights
  _Float16* wtEN[3] = {allocH(128 * 32), allocH(128 * 128), allocH(128 * 128)};
  _Float16* wtEE[3] = {allocH(128 * 32), allocH(128 * 128), allocH(128 * 128)};
  _Float16* wtDec[2] = {allocH(128 * 128), allocH(128 * 128)};
  _Float16* wtPE[NSTEPS][3];
  _Float16* wtPN[NSTEPS][3];
  for (int s = 0; s < NSTEPS; ++s) {
    wtPE[s][0] = allocH(128 * 384);
    wtPE[s][1] = allocH(128 * 128);
    wtPE[s][2] = allocH(128 * 128);
    wtPN[s][0] = allocH(128 * 256);
    wtPN[s][1] = allocH(128 * 128);
    wtPN[s][2] = allocH(128 * 128);
  }
  _Float16* nodeFeat = allocH((size_t)N_PART * 32);
  _Float16* edgeFeat = allocH((size_t)N_EDGE * 32);
  _Float16* vBuf = allocH((size_t)N_PART * LATENT);
  _Float16* eBuf = allocH((size_t)N_EDGE * LATENT);
  _Float16* h2Buf = allocH((size_t)N_PART * LATENT);
  float* aggBuf = allocF((size_t)N_PART * LATENT);

  auto T = [&](const float* W, _Float16* Wt, int K, int Kpad, int Nout) {
    int total = Nout * Kpad;
    transpose_w_kernel<<<(total + 255) / 256, 256, 0, stream>>>(W, Wt, K, Kpad,
                                                                Nout);
  };

  // ---- weight prep ---------------------------------------------------------
  T(F(I_EN_W0), wtEN[0], 30, 32, 128);
  T(F(I_EN_W1), wtEN[1], 128, 128, 128);
  T(F(I_EN_W2), wtEN[2], 128, 128, 128);
  T(F(I_EE_W0), wtEE[0], 3, 32, 128);
  T(F(I_EE_W1), wtEE[1], 128, 128, 128);
  T(F(I_EE_W2), wtEE[2], 128, 128, 128);
  T(F(I_DEC_W0), wtDec[0], 128, 128, 128);
  T(F(I_DEC_W1), wtDec[1], 128, 128, 128);
  for (int s = 0; s < NSTEPS; ++s) {
    int eb = I_PROC + 16 * s;
    int nb = eb + 8;
    T(F(eb + 0), wtPE[s][0], 384, 384, 128);
    T(F(eb + 1), wtPE[s][1], 128, 128, 128);
    T(F(eb + 2), wtPE[s][2], 128, 128, 128);
    T(F(nb + 0), wtPN[s][0], 256, 256, 128);
    T(F(nb + 1), wtPN[s][1], 128, 128, 128);
    T(F(nb + 2), wtPN[s][2], 128, 128, 128);
  }

  // ---- features ------------------------------------------------------------
  node_features_kernel<<<(N_PART + 255) / 256, 256, 0, stream>>>(
      pos, types, F(I_NODE_MEAN), F(I_NODE_STD), F(I_EMBED), nodeFeat, N_PART);
  edge_features_kernel<<<(N_EDGE + 255) / 256, 256, 0, stream>>>(
      pos, sidx, ridx, F(I_EDGE_MEAN), F(I_EDGE_STD), edgeFeat, N_EDGE);

  // ---- encoders ------------------------------------------------------------
  {
    MlpArgs a = {};
    a.X = nodeFeat;
    a.Wt1 = wtEN[0]; a.Wt2 = wtEN[1]; a.Wt3 = wtEN[2];
    a.b1 = F(I_EN_B0); a.b2 = F(I_EN_B1); a.b3 = F(I_EN_B2);
    a.ln_g = F(I_EN_LNG); a.ln_b = F(I_EN_LNB);
    a.out = vBuf;
    mlp_block_kernel<32, 3, 0><<<N_PART / 32, 256, 0, stream>>>(a);
  }
  {
    MlpArgs a = {};
    a.X = edgeFeat;
    a.Wt1 = wtEE[0]; a.Wt2 = wtEE[1]; a.Wt3 = wtEE[2];
    a.b1 = F(I_EE_B0); a.b2 = F(I_EE_B1); a.b3 = F(I_EE_B2);
    a.ln_g = F(I_EE_LNG); a.ln_b = F(I_EE_LNB);
    a.out = eBuf;
    mlp_block_kernel<32, 3, 1><<<N_EDGE / 32, 256, 0, stream>>>(a);
  }

  // ---- processor steps -----------------------------------------------------
  for (int s = 0; s < NSTEPS; ++s) {
    int eb = I_PROC + 16 * s;
    int nb = eb + 8;
    long na = (long)N_PART * LATENT;
    zero_f32_kernel<<<(int)((na + 255) / 256), 256, 0, stream>>>(aggBuf, na);

    MlpArgs ae = {};
    ae.Wt1 = wtPE[s][0]; ae.Wt2 = wtPE[s][1]; ae.Wt3 = wtPE[s][2];
    ae.b1 = F(eb + 3); ae.b2 = F(eb + 4); ae.b3 = F(eb + 5);
    ae.ln_b = F(eb + 6); ae.ln_g = F(eb + 7);
    ae.e_in = eBuf; ae.v_in = vBuf;
    ae.s_idx = sidx; ae.r_idx = ridx;
    ae.out = eBuf; ae.agg_out = aggBuf;
    mlp_block_kernel<384, 3, 2><<<N_EDGE / 32, 256, 0, stream>>>(ae);

    MlpArgs an = {};
    an.Wt1 = wtPN[s][0]; an.Wt2 = wtPN[s][1]; an.Wt3 = wtPN[s][2];
    an.b1 = F(nb + 3); an.b2 = F(nb + 4); an.b3 = F(nb + 5);
    an.ln_b = F(nb + 6); an.ln_g = F(nb + 7);
    an.v_in = vBuf; an.agg_in = aggBuf;
    an.out = vBuf;
    mlp_block_kernel<256, 3, 3><<<N_PART / 32, 256, 0, stream>>>(an);
  }

  // ---- decoder -------------------------------------------------------------
  {
    MlpArgs a = {};
    a.X = vBuf;
    a.Wt1 = wtDec[0]; a.Wt2 = wtDec[1];
    a.b1 = F(I_DEC_B0); a.b2 = F(I_DEC_B1);
    a.out = h2Buf;
    mlp_block_kernel<128, 2, 4><<<N_PART / 32, 256, 0, stream>>>(a);
  }
  dec_final_kernel<<<(N_PART + 255) / 256, 256, 0, stream>>>(
      h2Buf, F(I_DEC_W2), F(I_DEC_B2), F(I_OUT_STD), F(I_OUT_MEAN),
      (float*)d_out, N_PART);
}
